// TimeFreqRC_27333171871916
// MI455X (gfx1250) — matrix operand order repaired and checked by reading, not compiled, not run on hardware
//
#include <hip/hip_runtime.h>
#include <hip/hip_bf16.h>
#include <math.h>

// ---------------------------------------------------------------------------
// Echo-state reservoir scan, fused:  per step t (per 16-row batch group)
//   pre = [h_bf16 | x_t_bf16] (16 x 1152)  @  WcatT^T   (N=1024, K=1152)
//   h   = 0.7*h + 0.3*tanh(pre)            (f32 state in WMMA C-layout regs)
//   out[b,t,:] = h . Wout                  (LDS ds_add_f32 reduction)
// WcatT[n][k] = (k<1024 ? Wres[k][n] : Win[k-1024][n]) in bf16 (prep kernel).
// Grid: 8 workgroups x 512 threads (16 waves, 64 N-columns per wave).
// ---------------------------------------------------------------------------

typedef __attribute__((ext_vector_type(16))) __bf16          v16bf;
typedef __attribute__((ext_vector_type(8)))  float           v8f;
typedef __attribute__((ext_vector_type(8)))  unsigned short  ushort8x;

#define B_SZ    128
#define T_SZ    1024
#define IN_DIM  128
#define RES     1024
#define KTOT    (RES + IN_DIM)      // 1152
#define KTILES  (KTOT / 32)         // 36
#define LEAK    0.3f
#define ROWS    16                  // batch rows per workgroup
#define NWAVES  16
#define NPW     (RES / NWAVES)      // 64 columns per wave (4 WMMA N-tiles)
#define ASTRIDE (KTOT + 8)          // 1160 halves: 580 dwords % 64 = 4 -> no bank conflicts

static __device__ __forceinline__ unsigned short f2bf(float f) {
    unsigned int u = __float_as_uint(f);
    u += 0x7FFFu + ((u >> 16) & 1u);          // round-to-nearest-even
    return (unsigned short)(u >> 16);
}

// Build WcatT (bf16, [RES][KTOT] row-major) = [Wres ; Win] transposed.
__global__ void prep_wcat(const float* __restrict__ Win,
                          const float* __restrict__ Wres,
                          unsigned short* __restrict__ Wcat) {
    int idx = blockIdx.x * blockDim.x + threadIdx.x;
    if (idx >= RES * KTOT) return;
    int n = idx / KTOT;
    int k = idx - n * KTOT;
    float v = (k < RES) ? Wres[(size_t)k * RES + n]
                        : Win[(size_t)(k - RES) * RES + n];
    Wcat[idx] = f2bf(v);
}

__global__ __launch_bounds__(512)
void reservoir_scan(const float* __restrict__ x,
                    const unsigned short* __restrict__ Wcat,
                    const float* __restrict__ Wout,
                    float* __restrict__ out) {
    __shared__ __attribute__((aligned(16))) unsigned short As[ROWS * ASTRIDE];
    __shared__ float outacc[ROWS][2];

    const int tid    = threadIdx.x;
    const int wave   = tid >> 5;
    const int lane   = tid & 31;
    const int l15    = lane & 15;
    const int hi     = lane >> 4;        // which half of the wave
    const int bbase  = blockIdx.x * ROWS;
    const int nwbase = wave * NPW;

    // ---- init: zero A-tile LDS (h part) and out accumulator --------------
    for (int i = tid; i < ROWS * ASTRIDE; i += 512) As[i] = 0;
    if (tid < ROWS * 2) ((float*)outacc)[tid] = 0.f;

    // load x_0 into A-tile columns [RES, RES+IN_DIM)
    {
        const float* xp = x + (size_t)bbase * T_SZ * IN_DIM;
        for (int e = tid; e < ROWS * IN_DIM; e += 512) {
            int r = e >> 7, c = e & (IN_DIM - 1);
            As[r * ASTRIDE + RES + c] = f2bf(xp[(size_t)r * T_SZ * IN_DIM + c]);
        }
    }

    // persistent f32 reservoir state, in WMMA C layout:
    // hreg[t4][i] = h[m = i + 8*hi][n = nwbase + 16*t4 + l15]
    float hreg[4][8];
#pragma unroll
    for (int t4 = 0; t4 < 4; ++t4)
#pragma unroll
        for (int i = 0; i < 8; ++i) hreg[t4][i] = 0.f;

    // per-lane readout weights for this lane's 4 columns
    float w0[4], w1[4];
    const unsigned short* brow[4];
#pragma unroll
    for (int t4 = 0; t4 < 4; ++t4) {
        int n = nwbase + t4 * 16 + l15;
        w0[t4]   = Wout[n * 2 + 0];
        w1[t4]   = Wout[n * 2 + 1];
        brow[t4] = Wcat + (size_t)n * KTOT + hi * 8;  // B frag: K lo/hi interleave per half-wave (ISA table)
    }

    __syncthreads();

    for (int t = 0; t < T_SZ; ++t) {
        // ---------------- compute: pre = [h|x_t] @ WcatT^T ----------------
        v8f acc[4];
#pragma unroll
        for (int t4 = 0; t4 < 4; ++t4) acc[t4] = (v8f){};

        const unsigned short* arow = As + l15 * ASTRIDE + hi * 8;
#pragma unroll 4
        for (int kt = 0; kt < KTILES; ++kt) {
            const int kbase = kt * 32;
            union { v16bf v; ushort8x h[2]; } a;
            // A 16x32 bf16 frag: lane half selects K lo/hi interleave (ISA table)
            a.h[0] = *(const ushort8x*)(arow + kbase);
            a.h[1] = *(const ushort8x*)(arow + kbase + 16);
#pragma unroll
            for (int t4 = 0; t4 < 4; ++t4) {
                union { v16bf v; ushort8x h[2]; } b;
                b.h[0] = *(const ushort8x*)(brow[t4] + kbase);
                b.h[1] = *(const ushort8x*)(brow[t4] + kbase + 16);
                acc[t4] = __builtin_amdgcn_wmma_f32_16x16x32_bf16(
                    false, a.v, false, b.v, (short)0, acc[t4], false, false);
            }
        }

        // leaky-tanh state update (registers only)
#pragma unroll
        for (int t4 = 0; t4 < 4; ++t4)
#pragma unroll
            for (int i = 0; i < 8; ++i)
                hreg[t4][i] = (1.0f - LEAK) * hreg[t4][i] + LEAK * tanhf(acc[t4][i]);

        // per-lane readout partials (this lane's 4 columns)
        float p0[8], p1[8];
#pragma unroll
        for (int i = 0; i < 8; ++i) {
            float s0 = 0.f, s1 = 0.f;
#pragma unroll
            for (int t4 = 0; t4 < 4; ++t4) {
                s0 += hreg[t4][i] * w0[t4];
                s1 += hreg[t4][i] * w1[t4];
            }
            p0[i] = s0; p1[i] = s1;
        }

        __syncthreads();   // all waves done READING A-LDS (h_{t-1}, x_t)

        // ---------------- write phase ----------------
        // publish h_t (bf16) into A-LDS for next step's K
#pragma unroll
        for (int t4 = 0; t4 < 4; ++t4) {
            const int ncol = nwbase + t4 * 16 + l15;
#pragma unroll
            for (int i = 0; i < 8; ++i)
                As[(i + hi * 8) * ASTRIDE + ncol] = f2bf(hreg[t4][i]);
        }
        // readout reduction across waves
#pragma unroll
        for (int i = 0; i < 8; ++i) {
            const int m = i + hi * 8;
            atomicAdd(&outacc[m][0], p0[i]);
            atomicAdd(&outacc[m][1], p1[i]);
        }
        // stage x_{t+1}
        if (t + 1 < T_SZ) {
            const float* xp = x + ((size_t)bbase * T_SZ + (t + 1)) * IN_DIM;
            for (int e = tid; e < ROWS * IN_DIM; e += 512) {
                int r = e >> 7, c = e & (IN_DIM - 1);
                As[r * ASTRIDE + RES + c] = f2bf(xp[(size_t)r * T_SZ * IN_DIM + c]);
            }
        }

        __syncthreads();   // h_t / x_{t+1} / outacc visible

        // wave 0 drains the readout accumulator and re-zeros it
        if (wave == 0) {
            const int m = lane >> 1, j = lane & 1;
            out[(((size_t)(bbase + m)) * T_SZ + t) * 2 + j] = outacc[m][j];
            outacc[m][j] = 0.f;
        }
    }
}

extern "C" void kernel_launch(void* const* d_in, const int* in_sizes, int n_in,
                              void* d_out, int out_size, void* d_ws, size_t ws_size,
                              hipStream_t stream) {
    (void)in_sizes; (void)n_in; (void)out_size; (void)ws_size;
    const float* x    = (const float*)d_in[0];   // [128,1024,128]
    const float* Win  = (const float*)d_in[1];   // [128,1024]
    const float* Wres = (const float*)d_in[2];   // [1024,1024]
    const float* Wout = (const float*)d_in[3];   // [1024,2]
    float*       out  = (float*)d_out;           // [128,1024,2]

    unsigned short* Wcat = (unsigned short*)d_ws;   // 1024*1152*2 = 2.25 MB

    {
        int total = RES * KTOT;
        prep_wcat<<<(total + 255) / 256, 256, 0, stream>>>(Win, Wres, Wcat);
    }
    reservoir_scan<<<B_SZ / ROWS, NWAVES * 32, 0, stream>>>(x, Wcat, Wout, out);
}
